// GNN_3865470566846
// MI455X (gfx1250) — matrix operand order repaired and checked by reading, not compiled, not run on hardware
//
#include <hip/hip_runtime.h>
#include <hip/hip_bf16.h>

typedef __attribute__((ext_vector_type(16))) __bf16 v16bf;
typedef __attribute__((ext_vector_type(8)))  __bf16 v8bf;
typedef __attribute__((ext_vector_type(8)))  float  v8f;

// ---------------- degree / norm ----------------
__global__ void __launch_bounds__(256) k_init_deg(float* deg, int N) {
  int i = blockIdx.x * blockDim.x + threadIdx.x;
  if (i < N) deg[i] = 1.0f;                        // self-loop
}

__global__ void __launch_bounds__(256) k_count_deg(float* deg, const int* __restrict__ dst, int E) {
  int i = blockIdx.x * blockDim.x + threadIdx.x;
  if (i < E) unsafeAtomicAdd(&deg[dst[i]], 1.0f);
}

__global__ void __launch_bounds__(256) k_dinv(float* deg, int N) {
  int i = blockIdx.x * blockDim.x + threadIdx.x;
  if (i < N) deg[i] = rsqrtf(fmaxf(deg[i], 1.0f)); // deg -> dinv in place
}

// ---------------- bf16 staging ----------------
__global__ void __launch_bounds__(256)
k_cvt_pad(const float* __restrict__ X, __bf16* __restrict__ Xbf, long M, int K, int KP) {
  long i = (long)blockIdx.x * blockDim.x + threadIdx.x;
  if (i < M * KP) {
    int kp = (int)(i % KP);
    long m = i / KP;
    Xbf[i] = (kp < K) ? (__bf16)X[m * K + kp] : (__bf16)0.0f;
  }
}

// Wt[n][kp] = (kp < K) ? bf16(W[kp][n]) : 0   (transpose + pad)
__global__ void __launch_bounds__(256)
k_cvt_wt(const float* __restrict__ W, __bf16* __restrict__ Wt, int K, int N, int KP) {
  int i = blockIdx.x * blockDim.x + threadIdx.x;
  if (i < N * KP) {
    int kp = i % KP, n = i / KP;
    Wt[i] = (kp < K) ? (__bf16)W[(long)kp * N + n] : (__bf16)0.0f;
  }
}

// ---------------- WMMA GEMM ----------------
// act(A[M,K] @ Bt^T + bias). A bf16 row-major (K%32==0), Bt bf16 col-major.
// M multiple of 16. One wave: 16 rows x NT*16 cols; A fragment reused across tiles.
// Epilogue (compile-time selected): f32 C, bf16 Cbf, and/or agg = v * dinv[row]^2.
template<int K, int NT, bool BIASRELU, bool WF32, bool WBF, bool SL>
__global__ void __launch_bounds__(256)
k_gemm(const __bf16* __restrict__ A, const __bf16* __restrict__ Bt,
       const float* __restrict__ bias, float* __restrict__ C,
       __bf16* __restrict__ Cbf, float* __restrict__ agg,
       const float* __restrict__ dinv, int M)
{
  constexpr int N = NT * 16;
  const int lane  = threadIdx.x & 31;
  const int wave  = threadIdx.x >> 5;
  const int tileM = blockIdx.x * 8 + wave;
  if (tileM * 16 >= M) return;                     // wave-uniform (EXEC all-1 at WMMA)

  const int ml = lane & 15;                        // A row-in-tile / B,C col-in-tile
  const int kh = lane >> 4;                        // K-half selector
  const long rowBase = (long)(tileM * 16 + ml) * K;

  v8f acc[NT] = {};
#pragma unroll
  for (int k0 = 0; k0 < K; k0 += 32) {
    // A 16x32 bf16 fragment: two contiguous 8-element runs -> two b128 loads
    v8bf alo = *(const v8bf*)(A + rowBase + k0 + 8 * kh);
    v8bf ahi = *(const v8bf*)(A + rowBase + k0 + 16 + 8 * kh);
    v16bf a = __builtin_shufflevector(alo, ahi, 0,1,2,3,4,5,6,7,8,9,10,11,12,13,14,15);
#pragma unroll
    for (int t = 0; t < NT; ++t) {
      // B 32x16 fragment: two contiguous 8-element runs from transposed weights (two b128 loads)
      const int col = t * 16 + ml;
      v8bf blo = *(const v8bf*)(Bt + (long)col * K + k0 + 8 * kh);
      v8bf bhi = *(const v8bf*)(Bt + (long)col * K + k0 + 16 + 8 * kh);
      v16bf b = __builtin_shufflevector(blo, bhi, 0,1,2,3,4,5,6,7,8,9,10,11,12,13,14,15);
      acc[t] = __builtin_amdgcn_wmma_f32_16x16x32_bf16(false, a, false, b,
                                                       (short)0, acc[t], false, false);
    }
  }
#pragma unroll
  for (int r = 0; r < 8; ++r) {
    const int ro = tileM * 16 + r + 8 * kh;        // C/D layout: VGPR r -> M = r + 8*kh
    float dn2 = 0.0f;
    if (SL) { float dn = dinv[ro]; dn2 = dn * dn; }
#pragma unroll
    for (int t = 0; t < NT; ++t) {
      const int col = t * 16 + ml;
      float v = acc[t][r];
      if (BIASRELU) v = fmaxf(v + bias[col], 0.0f);
      if (WF32) C[(long)ro * N + col] = v;
      if (WBF)  Cbf[(long)ro * N + col] = (__bf16)v;
      if (SL)   agg[(long)ro * N + col] = v * dn2; // self-loop term of GCN aggregation
    }
  }
}

// ---------------- GCN aggregation: one wave per edge, 2 features/lane (H=64) ----------------
__global__ void __launch_bounds__(256)
k_edge_scatter(const float* __restrict__ hw, const int* __restrict__ src,
               const int* __restrict__ dst, const float* __restrict__ dinv,
               float* __restrict__ agg, int E, int H)
{
  long gid = (long)blockIdx.x * blockDim.x + threadIdx.x;
  int  e    = (int)(gid >> 5);
  int  lane = threadIdx.x & 31;
  if (e >= E) return;
  int s = src[e], d = dst[e];
  float nrm = dinv[s] * dinv[d];
  int f = lane * 2;
  float2 v = *(const float2*)(hw + (long)s * H + f);
  unsafeAtomicAdd(&agg[(long)d * H + f],     v.x * nrm);
  unsafeAtomicAdd(&agg[(long)d * H + f + 1], v.y * nrm);
}

// h = relu(agg + bias); emit f32 (for pooling) and/or bf16 (for next GEMM)
template<bool WF32, bool WBF>
__global__ void __launch_bounds__(256)
k_bias_relu(const float* __restrict__ agg, const float* __restrict__ bias,
            float* __restrict__ h, __bf16* __restrict__ hbf, long NH, int H)
{
  long i = (long)blockIdx.x * blockDim.x + threadIdx.x;
  if (i < NH) {
    float v = fmaxf(agg[i] + bias[i % H], 0.0f);
    if (WF32) h[i] = v;
    if (WBF)  hbf[i] = (__bf16)v;
  }
}

// ---------------- pooling ----------------
__global__ void __launch_bounds__(256) k_zero(float* p, long n) {
  long i = (long)blockIdx.x * blockDim.x + threadIdx.x;
  if (i < n) p[i] = 0.0f;
}

__global__ void __launch_bounds__(256)
k_pool_accum(const float* __restrict__ h, const int* __restrict__ batch,
             float* __restrict__ sum, float* __restrict__ cnt, long NH, int H)
{
  long i = (long)blockIdx.x * blockDim.x + threadIdx.x;
  if (i < NH) {
    int n = (int)(i / H), f = (int)(i % H);
    int g = batch[n];
    unsafeAtomicAdd(&sum[(long)g * H + f], h[i]);
    if (f == 0) unsafeAtomicAdd(&cnt[g], 1.0f);
  }
}

__global__ void __launch_bounds__(256)
k_pool_final(const float* __restrict__ sum, const float* __restrict__ cnt,
             __bf16* __restrict__ pooledbf, int G, int H)
{
  int i = blockIdx.x * blockDim.x + threadIdx.x;
  if (i < G * H) pooledbf[i] = (__bf16)(sum[i] / fmaxf(cnt[i / H], 1.0f));
}

// ---------------- head: out[g] = r2[g,:] . W3 + b3 (one wave per row) ----------------
__global__ void __launch_bounds__(256)
k_head(const float* __restrict__ r2, const float* __restrict__ W3,
       const float* __restrict__ b3, float* __restrict__ out, int G, int H2)
{
  int wave = (blockIdx.x * blockDim.x + threadIdx.x) >> 5;
  int lane = threadIdx.x & 31;
  if (wave >= G) return;
  float v = (lane < H2) ? r2[(long)wave * H2 + lane] * W3[lane] : 0.0f;
#pragma unroll
  for (int off = 16; off > 0; off >>= 1) v += __shfl_xor(v, off, 32);
  if (lane == 0) out[wave] = v + b3[0];
}

// ---------------- launch ----------------
extern "C" void kernel_launch(void* const* d_in, const int* in_sizes, int n_in,
                              void* d_out, int out_size, void* d_ws, size_t ws_size,
                              hipStream_t stream) {
  const float* x      = (const float*)d_in[0];
  const int*   ei     = (const int*)  d_in[1];
  const int*   batch  = (const int*)  d_in[2];
  const float* W_emb  = (const float*)d_in[3];
  const float* b_emb  = (const float*)d_in[4];
  const float* conv_W = (const float*)d_in[5];
  const float* conv_b = (const float*)d_in[6];
  const float* W1     = (const float*)d_in[7];
  const float* b1     = (const float*)d_in[8];
  const float* W2     = (const float*)d_in[9];
  const float* b2     = (const float*)d_in[10];
  const float* W3     = (const float*)d_in[11];
  const float* b3     = (const float*)d_in[12];
  float* out = (float*)d_out;

  const int H   = in_sizes[4];            // 64
  const int FI  = in_sizes[3] / H;        // 100
  const int N   = in_sizes[0] / FI;       // 100000
  const int E   = in_sizes[1] / 2;        // 1600000
  const int L   = in_sizes[5] / (H * H);  // 3
  const int G   = out_size;               // 256
  const int H2  = in_sizes[10];           // 32
  const int KPX = 128;                    // F_IN padded to multiple of 32

  // workspace carve-up (256B aligned segments)
  char* wp = (char*)d_ws;
  auto alloc = [&](size_t bytes) { void* r = wp; wp += (bytes + 255) & ~(size_t)255; return r; };
  float*  deg      = (float*) alloc((size_t)N * 4);
  float*  h        = (float*) alloc((size_t)N * H * 4);
  float*  hw       = (float*) alloc((size_t)N * H * 4);
  float*  agg      = (float*) alloc((size_t)N * H * 4);
  __bf16* hbf      = (__bf16*)alloc((size_t)N * H * 2);
  __bf16* xbf      = (__bf16*)alloc((size_t)N * KPX * 2);
  __bf16* wtE      = (__bf16*)alloc((size_t)H * KPX * 2);       // W_emb^T padded
  __bf16* wtC      = (__bf16*)alloc((size_t)L * H * H * 2);     // conv_W^T
  __bf16* wt1      = (__bf16*)alloc((size_t)H * H * 2);
  __bf16* wt2      = (__bf16*)alloc((size_t)H2 * H * 2);
  float*  sum      = (float*) alloc((size_t)G * H * 4);
  float*  cnt      = (float*) alloc((size_t)G * 4);
  __bf16* pooledbf = (__bf16*)alloc((size_t)G * H * 2);
  __bf16* r1bf     = (__bf16*)alloc((size_t)G * H * 2);
  float*  r2       = (float*) alloc((size_t)G * H2 * 4);

  auto cdiv = [](long a, long b) { return (int)((a + b - 1) / b); };
  const long NH = (long)N * H;
  const int* src = ei;
  const int* dst = ei + E;

  // 1) normalization coefficients
  k_init_deg <<<cdiv(N, 256), 256, 0, stream>>>(deg, N);
  k_count_deg<<<cdiv(E, 256), 256, 0, stream>>>(deg, dst, E);
  k_dinv     <<<cdiv(N, 256), 256, 0, stream>>>(deg, N);

  // 2) bf16 staging: x (padded) + all weights (transposed)
  k_cvt_pad<<<cdiv((long)N * KPX, 256), 256, 0, stream>>>(x, xbf, N, FI, KPX);
  k_cvt_wt <<<cdiv(H * KPX, 256), 256, 0, stream>>>(W_emb, wtE, FI, H, KPX);
  for (int l = 0; l < L; ++l)
    k_cvt_wt<<<cdiv(H * H, 256), 256, 0, stream>>>(conv_W + (size_t)l * H * H,
                                                   wtC + (size_t)l * H * H, H, H, H);
  k_cvt_wt<<<cdiv(H * H, 256), 256, 0, stream>>>(W1, wt1, H, H, H);
  k_cvt_wt<<<cdiv(H2 * H, 256), 256, 0, stream>>>(W2, wt2, H, H2, H);

  // 3) embed: hbf = bf16(relu(x @ W_emb + b_emb))   (f32 copy never consumed)
  k_gemm<128, 4, true, false, true, false><<<cdiv(N / 16, 8), 256, 0, stream>>>(
      xbf, wtE, b_emb, nullptr, hbf, nullptr, nullptr, N);

  // 4) stacked GCNConv: GEMM writes hw (f32) and agg = hw*dinv^2 (fused self-loop)
  for (int l = 0; l < L; ++l) {
    k_gemm<64, 4, false, true, false, true><<<cdiv(N / 16, 8), 256, 0, stream>>>(
        hbf, wtC + (size_t)l * H * H, nullptr, hw, nullptr, agg, deg, N);
    k_edge_scatter<<<cdiv((long)E * 32, 256), 256, 0, stream>>>(hw, src, dst, deg, agg, E, H);
    if (l + 1 < L)  // intermediate: only bf16 needed (next GEMM)
      k_bias_relu<false, true><<<cdiv(NH, 256), 256, 0, stream>>>(
          agg, conv_b + (size_t)l * H, nullptr, hbf, NH, H);
    else            // last: only f32 needed (pooling)
      k_bias_relu<true, false><<<cdiv(NH, 256), 256, 0, stream>>>(
          agg, conv_b + (size_t)l * H, h, nullptr, NH, H);
  }

  // 5) global mean pool (sum and cnt contiguous -> single zero pass)
  k_zero      <<<cdiv((long)G * H + G, 256), 256, 0, stream>>>(sum, (long)G * H + G);
  k_pool_accum<<<cdiv(NH, 256), 256, 0, stream>>>(h, batch, sum, cnt, NH, H);
  k_pool_final<<<cdiv((long)G * H, 256), 256, 0, stream>>>(sum, cnt, pooledbf, G, H);

  // 6) MLP readout
  k_gemm<64, 4, true, false, true, false><<<cdiv(G / 16, 8), 256, 0, stream>>>(
      pooledbf, wt1, b1, nullptr, r1bf, nullptr, nullptr, G);
  k_gemm<64, 2, true, true, false, false><<<cdiv(G / 16, 8), 256, 0, stream>>>(
      r1bf, wt2, b2, r2, nullptr, nullptr, nullptr, G);
  k_head<<<cdiv((long)G * 32, 256), 256, 0, stream>>>(r2, W3, b3, out, G, H2);
}
